// RadianceRenderer_23132693856652
// MI455X (gfx1250) — matrix operand order repaired and checked by reading, not compiled, not run on hardware
//
#include <hip/hip_runtime.h>

// Radiance field renderer, fused per-ray kernel for gfx1250 (MI455X).
// Dominant compute (1M x 128 x 128 GEMM) runs on v_wmma_f32_16x16x32_f16,
// with both A (activations) and B (W2^T) fragments served from LDS.

#define N_RAYS   8192
#define T_SAMP   128
#define HID      128
#define SH_DIM   16
#define RAY_EXT  10.0f
#define P_SAMP   32      // samples per pass (4 passes of 32)

typedef _Float16 half_t;
typedef __attribute__((ext_vector_type(16))) _Float16 v16h;
typedef __attribute__((ext_vector_type(8)))  _Float16 v8h;
typedef __attribute__((ext_vector_type(8)))  float    v8f;

// ---------------------------------------------------------------------------
// Prep: W2 (f32, [k][n]) -> W2^T f16 [n][k] so WMMA B-fragments are contiguous.
// ---------------------------------------------------------------------------
__global__ void w2t_pack_kernel(const float* __restrict__ W2,
                                half_t* __restrict__ w2t) {
  int i = blockIdx.x * blockDim.x + threadIdx.x;
  if (i < HID * HID) {
    int k = i >> 7, n = i & 127;
    w2t[n * HID + k] = (half_t)W2[k * HID + n];
  }
}

// A fragment: 16x32 f16 tile of H1 (row-major in LDS, row stride HID halves).
// ISA layout: lanes 0-15 hold M=lane, K = kb+{0..7} (v[0..7]) and kb+16+{0..7}
// (v[8..15]); lanes 16-31 hold the +8 K offsets.
__device__ __forceinline__ v16h load_a_frag(const half_t* base,
                                            int mbase, int kb, int lane) {
  int row = mbase + (lane & 15);
  int hi  = lane >> 4;
  const half_t* p = base + row * HID + kb + hi * 8;
  v8h lo = *(const v8h*)p;
  v8h up = *(const v8h*)(p + 16);
  v16h a;
#pragma unroll
  for (int i = 0; i < 8; ++i) { a[i] = lo[i]; a[i + 8] = up[i]; }
  return a;
}

// B fragment: 32x16 f16 tile from W2^T [n][k] in LDS.
// ISA layout: lanes 0-15 hold K=kb+{0..7} (v[0..7]) and kb+16+{0..7} (v[8..15])
// of column N=lane; lanes 16-31 hold the +8 K offsets.
__device__ __forceinline__ v16h load_b_frag(const half_t* w2t,
                                            int nb, int kb, int lane) {
  int col = nb + (lane & 15);
  int hi  = lane >> 4;
  const half_t* p = w2t + col * HID + kb + hi * 8;
  v8h lo = *(const v8h*)p;
  v8h up = *(const v8h*)(p + 16);
  v16h b;
#pragma unroll
  for (int i = 0; i < 8; ++i) { b[i] = lo[i]; b[i + 8] = up[i]; }
  return b;
}

// ---------------------------------------------------------------------------
// Fused per-ray renderer: 1 block (4 wave32) per ray.
// ---------------------------------------------------------------------------
__global__ __launch_bounds__(128)
void render_kernel(const float* __restrict__ origins,
                   const float* __restrict__ dirs,
                   const float* __restrict__ u,
                   const float* __restrict__ W1,
                   const float* __restrict__ b1,
                   const float* __restrict__ b2,
                   const float* __restrict__ Wd,
                   const float* __restrict__ bd,
                   const float* __restrict__ Wc,
                   const float* __restrict__ bc,
                   const half_t* __restrict__ w2t,
                   float* __restrict__ out) {
  __shared__ half_t sh_w2t[HID * HID];               // 32 KB, block-resident B
  __shared__ half_t sh_h1[P_SAMP * HID];             // 8 KB, f16 A tiles
  __shared__ half_t sh_h2[P_SAMP * HID];             // 8 KB, f16 H2
  __shared__ float  sh_w1[3 * HID];
  __shared__ float  sh_b1[HID];
  __shared__ float  sh_b2[HID];
  __shared__ float  sh_wd[HID];
  __shared__ float  sh_wc[(HID + SH_DIM) * 3];
  __shared__ float  sh_sig[T_SAMP];
  __shared__ float  sh_col[T_SAMP * 3];
  __shared__ float  sh_scan[T_SAMP];
  __shared__ float  sh_red[T_SAMP * 4];
  __shared__ float  sh_part[P_SAMP * 16];            // head partial dots

  const int t    = threadIdx.x;
  const int r    = blockIdx.x;
  const int lane = t & 31;
  const int wave = t >> 5;

  // ---- stage weights into LDS (W2^T via wide b128 copies) ----
  sh_w1[t]       = W1[t];
  sh_w1[t + 128] = W1[t + 128];
  sh_w1[t + 256] = W1[t + 256];
  sh_b1[t] = b1[t];
  sh_b2[t] = b2[t];
  sh_wd[t] = Wd[t];
  for (int i = t; i < (HID + SH_DIM) * 3; i += 128) sh_wc[i] = Wc[i];
#pragma unroll 1
  for (int i = t * 8; i < HID * HID; i += 128 * 8) {
    *(v8h*)(sh_w2t + i) = *(const v8h*)(w2t + i);
  }

  // ---- per-ray setup (redundant per thread; all values needed per lane) ----
  float ox = origins[r * 3 + 0], oy = origins[r * 3 + 1], oz = origins[r * 3 + 2];
  float dx = dirs[r * 3 + 0],    dy = dirs[r * 3 + 1],    dz = dirs[r * 3 + 2];
  float ixd = 1.0f / dx, iyd = 1.0f / dy, izd = 1.0f / dz;
  float t1x = (-1.0f - ox) * ixd, t2x = (1.0f - ox) * ixd;
  float t1y = (-1.0f - oy) * iyd, t2y = (1.0f - oy) * iyd;
  float t1z = (-1.0f - oz) * izd, t2z = (1.0f - oz) * izd;
  float tnear = fmaxf(fmaxf(fminf(t1x, t2x), fminf(t1y, t2y)), fminf(t1z, t2z));
  tnear = fmaxf(tnear, 0.0f);
  float tfar = fminf(fminf(fmaxf(t1x, t2x), fmaxf(t1y, t2y)), fmaxf(t1z, t2z));
  bool  active = tfar > tnear;
  float tfarc  = fmaxf(tfar, tnear + 1e-3f);
  float dnorm  = sqrtf(dx * dx + dy * dy + dz * dz);
  float nx = dx / dnorm, ny = dy / dnorm, nz = dz / dnorm;

  // real SH degree 3 (16 coeffs)
  float x2 = nx * nx, y2 = ny * ny, z2 = nz * nz;
  float xy = nx * ny, yzp = ny * nz, xz = nx * nz;
  float ynm[SH_DIM];
  ynm[0]  = 0.282094791773878f;
  ynm[1]  = -0.48860251190292f * ny;
  ynm[2]  = 0.48860251190292f * nz;
  ynm[3]  = -0.48860251190292f * nx;
  ynm[4]  = 1.0925484305920792f * xy;
  ynm[5]  = -1.0925484305920792f * yzp;
  ynm[6]  = 0.94617469575756f * z2 - 0.31539156525252f;
  ynm[7]  = -1.0925484305920792f * xz;
  ynm[8]  = 0.5462742152960396f * (x2 - y2);
  ynm[9]  = 0.5900435899266435f * ny * (-3.0f * x2 + y2);
  ynm[10] = 2.8906114426405538f * xy * nz;
  ynm[11] = 0.4570457994644658f * ny * (1.0f - 5.0f * z2);
  ynm[12] = 0.3731763325901154f * nz * (5.0f * z2 - 3.0f);
  ynm[13] = 0.4570457994644658f * nx * (1.0f - 5.0f * z2);
  ynm[14] = 1.445305721320277f * nz * (x2 - y2);
  ynm[15] = 0.5900435899266435f * nx * (-x2 + 3.0f * y2);

  const float invT = 1.0f / (float)T_SAMP;
  const float dt   = tfarc - tnear;

  __syncthreads();

  // ================= four passes of 32 samples =================
#pragma unroll 1
  for (int pass = 0; pass < 4; ++pass) {
    const int sl = t & 31;        // local sample row (0..31)
    const int q  = t >> 5;        // quarter of the hidden dim
    const int s  = pass * P_SAMP + sl;

    float uv = u[r * T_SAMP + s];
    float ts = tnear + dt * ((float)s + uv) * invT;
    float px = ox + dx * ts, py = oy + dy * ts, pz = oz + dz * ts;

    // ---- layer 1 (K=3, VALU) -> f16 H1 in LDS; 4 threads per sample ----
    for (int j = q * 32; j < q * 32 + 32; ++j) {
      float h = sh_b1[j] + px * sh_w1[j] + py * sh_w1[HID + j] + pz * sh_w1[2 * HID + j];
      sh_h1[sl * HID + j] = (half_t)fmaxf(h, 0.0f);
    }
    __syncthreads();

    // ---- layer 2 via WMMA: wave -> (M-tile, N-range); A frags hoisted ----
    const int mbase  = (wave & 1) * 16;
    const int nstart = (wave >> 1) * 64;
    v16h a0 = load_a_frag(sh_h1, mbase, 0,  lane);
    v16h a1 = load_a_frag(sh_h1, mbase, 32, lane);
    v16h a2 = load_a_frag(sh_h1, mbase, 64, lane);
    v16h a3 = load_a_frag(sh_h1, mbase, 96, lane);
#pragma unroll 1
    for (int nb = nstart; nb < nstart + 64; nb += 32) {
      v8f acc0 = {}, acc1 = {};
      v16h bb0, bb1;
      bb0 = load_b_frag(sh_w2t, nb,      0,  lane);
      bb1 = load_b_frag(sh_w2t, nb + 16, 0,  lane);
      acc0 = __builtin_amdgcn_wmma_f32_16x16x32_f16(false, a0, false, bb0, (short)0, acc0, false, false);
      acc1 = __builtin_amdgcn_wmma_f32_16x16x32_f16(false, a0, false, bb1, (short)0, acc1, false, false);
      bb0 = load_b_frag(sh_w2t, nb,      32, lane);
      bb1 = load_b_frag(sh_w2t, nb + 16, 32, lane);
      acc0 = __builtin_amdgcn_wmma_f32_16x16x32_f16(false, a1, false, bb0, (short)0, acc0, false, false);
      acc1 = __builtin_amdgcn_wmma_f32_16x16x32_f16(false, a1, false, bb1, (short)0, acc1, false, false);
      bb0 = load_b_frag(sh_w2t, nb,      64, lane);
      bb1 = load_b_frag(sh_w2t, nb + 16, 64, lane);
      acc0 = __builtin_amdgcn_wmma_f32_16x16x32_f16(false, a2, false, bb0, (short)0, acc0, false, false);
      acc1 = __builtin_amdgcn_wmma_f32_16x16x32_f16(false, a2, false, bb1, (short)0, acc1, false, false);
      bb0 = load_b_frag(sh_w2t, nb,      96, lane);
      bb1 = load_b_frag(sh_w2t, nb + 16, 96, lane);
      acc0 = __builtin_amdgcn_wmma_f32_16x16x32_f16(false, a3, false, bb0, (short)0, acc0, false, false);
      acc1 = __builtin_amdgcn_wmma_f32_16x16x32_f16(false, a3, false, bb1, (short)0, acc1, false, false);

      // fuse bias + ReLU, scatter both D tiles per ISA layout
      const int hi = lane >> 4;
      {
        const int nc = nb + (lane & 15);
        const float bias = sh_b2[nc];
#pragma unroll
        for (int rr = 0; rr < 8; ++rr) {
          float v = acc0[rr] + bias;
          sh_h2[(mbase + rr + hi * 8) * HID + nc] = (half_t)fmaxf(v, 0.0f);
        }
      }
      {
        const int nc = nb + 16 + (lane & 15);
        const float bias = sh_b2[nc];
#pragma unroll
        for (int rr = 0; rr < 8; ++rr) {
          float v = acc1[rr] + bias;
          sh_h2[(mbase + rr + hi * 8) * HID + nc] = (half_t)fmaxf(v, 0.0f);
        }
      }
    }
    __syncthreads();

    // ---- heads: 4 threads per sample compute 32-wide partial dots ----
    {
      const half_t* hrow = sh_h2 + sl * HID;
      float p0 = 0.0f, p1 = 0.0f, p2 = 0.0f, p3 = 0.0f;
      for (int j = q * 32; j < q * 32 + 32; ++j) {
        float hv = (float)hrow[j];
        p0 += hv * sh_wd[j];
        p1 += hv * sh_wc[j * 3 + 0];
        p2 += hv * sh_wc[j * 3 + 1];
        p3 += hv * sh_wc[j * 3 + 2];
      }
      if (q == 0) {
        p0 += bd[0]; p1 += bc[0]; p2 += bc[1]; p3 += bc[2];
#pragma unroll
        for (int k = 0; k < SH_DIM; ++k) {
          float yv = ynm[k];
          p1 += yv * sh_wc[(HID + k) * 3 + 0];
          p2 += yv * sh_wc[(HID + k) * 3 + 1];
          p3 += yv * sh_wc[(HID + k) * 3 + 2];
        }
      }
      sh_part[sl * 16 + q * 4 + 0] = p0;
      sh_part[sl * 16 + q * 4 + 1] = p1;
      sh_part[sl * 16 + q * 4 + 2] = p2;
      sh_part[sl * 16 + q * 4 + 3] = p3;
    }
    __syncthreads();
    if (t < P_SAMP) {
      const float* pp = sh_part + t * 16;
      float sacc = pp[0] + pp[4] + pp[8]  + pp[12];
      float c0   = pp[1] + pp[5] + pp[9]  + pp[13];
      float c1   = pp[2] + pp[6] + pp[10] + pp[14];
      float c2   = pp[3] + pp[7] + pp[11] + pp[15];
      bool mask = active && fabsf(px) <= 1.0f && fabsf(py) <= 1.0f && fabsf(pz) <= 1.0f;
      float sig = (sacc > 20.0f) ? sacc : log1pf(expf(sacc));
      float m = mask ? 1.0f : 0.0f;
      sh_sig[s]         = sig * m;
      sh_col[s * 3 + 0] = m / (1.0f + expf(-c0));
      sh_col[s * 3 + 1] = m / (1.0f + expf(-c1));
      sh_col[s * 3 + 2] = m / (1.0f + expf(-c2));
    }
    __syncthreads();
  }

  // ================= transmittance integration =================
  float uT  = u[r * T_SAMP + t];
  float tsT = tnear + dt * ((float)t + uT) * invT;
  float tnext;
  if (t < T_SAMP - 1) {
    float un = u[r * T_SAMP + t + 1];
    tnext = tnear + dt * ((float)(t + 1) + un) * invT;
  } else {
    tnext = tfarc * RAY_EXT;
  }
  float delta = tnext - tsT;
  float sd = sh_sig[t] * delta * dnorm;

  // inclusive scan over 128 samples (Hillis-Steele in LDS)
  float xacc = sd;
  sh_scan[t] = xacc;
  __syncthreads();
  for (int off = 1; off < T_SAMP; off <<= 1) {
    float v = (t >= off) ? sh_scan[t - off] : 0.0f;
    __syncthreads();
    xacc += v;
    sh_scan[t] = xacc;
    __syncthreads();
  }
  float csum = xacc;
  float w = expf(sd - csum) - expf(-csum);

  sh_red[t * 4 + 0] = w * sh_col[t * 3 + 0];
  sh_red[t * 4 + 1] = w * sh_col[t * 3 + 1];
  sh_red[t * 4 + 2] = w * sh_col[t * 3 + 2];
  sh_red[t * 4 + 3] = w;
  __syncthreads();
  for (int sred = 64; sred >= 1; sred >>= 1) {
    if (t < sred) {
#pragma unroll
      for (int c = 0; c < 4; ++c) sh_red[t * 4 + c] += sh_red[(t + sred) * 4 + c];
    }
    __syncthreads();
  }
  if (t == 0) {
#pragma unroll
    for (int c = 0; c < 4; ++c) out[r * 4 + c] = active ? sh_red[c] : 0.0f;
  }
}

// ---------------------------------------------------------------------------
// Launch wrapper
// ---------------------------------------------------------------------------
extern "C" void kernel_launch(void* const* d_in, const int* in_sizes, int n_in,
                              void* d_out, int out_size, void* d_ws, size_t ws_size,
                              hipStream_t stream) {
  (void)in_sizes; (void)n_in; (void)out_size; (void)ws_size;
  const float* origins = (const float*)d_in[0];
  const float* dirs    = (const float*)d_in[1];
  const float* u       = (const float*)d_in[2];
  const float* W1      = (const float*)d_in[3];
  const float* b1      = (const float*)d_in[4];
  const float* W2      = (const float*)d_in[5];
  const float* b2      = (const float*)d_in[6];
  const float* Wd      = (const float*)d_in[7];
  const float* bd      = (const float*)d_in[8];
  const float* Wc      = (const float*)d_in[9];
  const float* bc      = (const float*)d_in[10];
  float* out = (float*)d_out;

  half_t* w2t = (half_t*)d_ws;  // 128*128 f16 = 32 KB

  w2t_pack_kernel<<<(HID * HID + 255) / 256, 256, 0, stream>>>(W2, w2t);
  render_kernel<<<N_RAYS, 128, 0, stream>>>(origins, dirs, u, W1, b1, b2,
                                            Wd, bd, Wc, bc, w2t, out);
}
